// PanguModel_56968446214497
// MI455X (gfx1250) — matrix operand order repaired and checked by reading, not compiled, not run on hardware
//
#include <hip/hip_runtime.h>
#include <hip/hip_bf16.h>
#include <math.h>

// ---------------------------------------------------------------------------
// Pangu 3D-Swin block for MI455X (gfx1250): bf16 WMMA + async LDS pipeline.
// ---------------------------------------------------------------------------

#define TOK    69120      // Z*H*W tokens
#define CNL    384        // channels
#define NWIN   576        // windows
#define NTOK   120        // tokens per window
#define NHEAD  12
#define DHEAD  32

typedef __attribute__((ext_vector_type(16))) __bf16 bf16x16;
typedef __attribute__((ext_vector_type(8)))  float  f32x8;
typedef int i32x4 __attribute__((vector_size(16)));

union FragU {
    unsigned u[8];
    bf16x16  v;
};

union Vec8U {
    uint4          q;
    unsigned short s[8];
};

// ---- gfx1250 async global->LDS path (guarded so either toolchain compiles)
#if defined(__AMDGCN__) && __has_builtin(__builtin_amdgcn_global_load_async_to_lds_b128)
#define HAVE_ASYNC_LDS 1
#define GPTR4(p) ((__attribute__((address_space(1))) i32x4*)(p))
#define LPTR4(p) ((__attribute__((address_space(3))) i32x4*)(p))
#if __has_builtin(__builtin_amdgcn_s_wait_asynccnt)
#define ASYNC_WAIT() __builtin_amdgcn_s_wait_asynccnt(0)
#else
#define ASYNC_WAIT() asm volatile("s_wait_asynccnt 0x0" ::: "memory")
#endif
#else
#define HAVE_ASYNC_LDS 0
#define ASYNC_WAIT() ((void)0)
#endif

static __device__ __forceinline__ unsigned short f2bf(float f) {
    unsigned u = __builtin_bit_cast(unsigned, f);
    unsigned r = u + 0x7FFFu + ((u >> 16) & 1u);
    return (unsigned short)(r >> 16);
}

// window-ordered row -> original token
static __device__ __forceinline__ int win_to_tok(int r) {
    int wi = r / NTOK, p = r % NTOK;
    int zb = wi / 144, hb = (wi / 18) % 8, wb = wi % 18;
    int zi = p / 60, hi = (p / 10) % 6, wi2 = p % 10;
    int z = zb * 2 + zi, hh = hb * 6 + hi, ww = wb * 10 + wi2;
    return (z * 48 + hh) * 180 + ww;
}

// original token -> window-ordered row
static __device__ __forceinline__ int tok_to_win(int t) {
    int z = t / 8640;          // 48*180
    int rem = t % 8640;
    int hh = rem / 180, ww = rem % 180;
    int zb = z >> 1, zi = z & 1;
    int hb = hh / 6, hi = hh % 6;
    int wb = ww / 10, wi2 = ww % 10;
    int wi = (zb * 8 + hb) * 18 + wb;
    int p = (zi * 6 + hi) * 10 + wi2;
    return wi * NTOK + p;
}

// relative position index, mirrors _position_index(WS)
static __device__ __forceinline__ int pos_idx(int n, int m) {
    int zn = n / 60, hn = (n / 10) % 6, wn = n % 10;
    int zm = m / 60, hm = (m / 10) % 6, wm = m % 10;
    return (zn + 2 * zm) * 684 + (hn + 6 * hm) * 19 + (wn - wm + 9);
}

// 32-byte global->LDS copy (async when available)
static __device__ __forceinline__ void copy32(const unsigned short* src,
                                              unsigned short* dst) {
#if HAVE_ASYNC_LDS
    __builtin_amdgcn_global_load_async_to_lds_b128(GPTR4(src), LPTR4(dst), 0, 0);
    __builtin_amdgcn_global_load_async_to_lds_b128(GPTR4(src + 8), LPTR4(dst + 8), 0, 0);
#else
    uint4 d0 = *(const uint4*)(src);
    uint4 d1 = *(const uint4*)(src + 8);
    *(uint4*)dst       = d0;
    *(uint4*)(dst + 8) = d1;
#endif
}

// ---------------------------------------------------------------------------
// elementwise fp32 -> bf16
__global__ void cvt_bf16_kernel(const float* __restrict__ s,
                                unsigned short* __restrict__ d, long n) {
    long i = (long)blockIdx.x * blockDim.x + threadIdx.x;
    long stride = (long)gridDim.x * blockDim.x;
    for (; i < n; i += stride) d[i] = f2bf(s[i]);
}

// build window-ordered bf16 copy of x
__global__ void make_xw_kernel(const float* __restrict__ x,
                               unsigned short* __restrict__ xw) {
    long total = (long)TOK * CNL;
    long i = (long)blockIdx.x * blockDim.x + threadIdx.x;
    long stride = (long)gridDim.x * blockDim.x;
    for (; i < total; i += stride) {
        int r = (int)(i / CNL), c = (int)(i % CNL);
        int t = win_to_tok(r);
        xw[i] = f2bf(x[(long)t * CNL + c]);
    }
}

// ---------------------------------------------------------------------------
// Generic bf16 GEMM: C[M,N] = A[M,K] * B[K,N] + bias, tile 128x64, 8 waves,
// double-buffered LDS. A staged via async global->LDS; B staged with coalesced
// 16B row loads + transposed LDS stores.
// mode 0: QKV scatter (q scaled by 1/sqrt(32)) -> outB (q/k/v segments)
// mode 1: fp32 row-major -> outF
// mode 2: exact GeLU -> bf16 row-major -> outB
// mode 3: fp32 row-major -> outF
__global__ __launch_bounds__(256)
void gemm_bf16_kernel(const unsigned short* __restrict__ A,
                      const unsigned short* __restrict__ B,
                      const float* __restrict__ bias,
                      float* __restrict__ outF,
                      unsigned short* __restrict__ outB,
                      int M, int N, int K, int mode) {
    const int tid  = threadIdx.x;
    const int lane = tid & 31;
    const int wv   = tid >> 5;
    const int wm   = wv & 3;       // 4 wave rows
    const int wn   = wv >> 2;      // 2 wave cols
    const int m0   = blockIdx.x * 128;
    const int n0   = blockIdx.y * 64;
    const int half = lane >> 4;
    const int l15  = lane & 15;

    __shared__ unsigned short a_lds[2][128 * 32];
    __shared__ unsigned short b_lds[2][64 * 32];

    const int arow = tid >> 1, ah = tid & 1;       // A staging: 32B / thread
    const int bkk = tid >> 3;                      // B staging: K row 0..31
    const int bcg = (tid & 7) * 8;                 // B staging: col group

    f32x8 acc[2][2] = {};

    // prologue: stage k-tile 0 into buffer 0
    copy32(A + (size_t)(m0 + arow) * K + ah * 16, &a_lds[0][arow * 32 + ah * 16]);
    {
        Vec8U u;
        u.q = *(const uint4*)(B + (size_t)bkk * N + n0 + bcg);
#pragma unroll
        for (int i = 0; i < 8; ++i)
            b_lds[0][(bcg + i) * 32 + bkk] = u.s[i];
    }
    ASYNC_WAIT();
    __syncthreads();

    const int nk = K >> 5;
    for (int ki = 0; ki < nk; ++ki) {
        const int buf = ki & 1;
        // stage next k-tile into other buffer while computing this one
        if (ki + 1 < nk) {
            int k1 = (ki + 1) << 5;
            copy32(A + (size_t)(m0 + arow) * K + k1 + ah * 16,
                   &a_lds[buf ^ 1][arow * 32 + ah * 16]);
            Vec8U u;
            u.q = *(const uint4*)(B + (size_t)(k1 + bkk) * N + n0 + bcg);
#pragma unroll
            for (int i = 0; i < 8; ++i)
                b_lds[buf ^ 1][(bcg + i) * 32 + bkk] = u.s[i];
        }
        if (ki + 2 < nk) {  // pull k-tile+2 toward the caches
            int k2 = (ki + 2) << 5;
            __builtin_prefetch(A + (size_t)(m0 + arow) * K + k2 + ah * 16, 0, 1);
            __builtin_prefetch(B + (size_t)(k2 + bkk) * N + n0 + bcg, 0, 1);
        }

        FragU afr[2], bfr[2];
#pragma unroll
        for (int i = 0; i < 2; ++i) {
            int row = wm * 32 + i * 16 + l15;
#pragma unroll
            for (int v = 0; v < 8; ++v) {
                int kk = half * 16 + 2 * v;
                afr[i].u[v] = *(const unsigned*)&a_lds[buf][row * 32 + kk];
            }
        }
#pragma unroll
        for (int j = 0; j < 2; ++j) {
            int col = wn * 32 + j * 16 + l15;
#pragma unroll
            for (int v = 0; v < 8; ++v)
                bfr[j].u[v] = *(const unsigned*)&b_lds[buf][col * 32 + half * 16 + 2 * v];
        }
#pragma unroll
        for (int i = 0; i < 2; ++i)
#pragma unroll
            for (int j = 0; j < 2; ++j)
                acc[i][j] = __builtin_amdgcn_wmma_f32_16x16x32_bf16(
                    false, afr[i].v, false, bfr[j].v, (short)0, acc[i][j],
                    false, false);

        ASYNC_WAIT();
        __syncthreads();
    }

    const float QSCALE = 0.17677669529663687f;  // 1/sqrt(32)
#pragma unroll
    for (int i = 0; i < 2; ++i) {
#pragma unroll
        for (int j = 0; j < 2; ++j) {
#pragma unroll
            for (int v = 0; v < 8; ++v) {
                int row = m0 + wm * 32 + i * 16 + v + half * 8;
                int col = n0 + wn * 32 + j * 16 + l15;
                float val = acc[i][j][v] + (bias ? bias[col] : 0.0f);
                if (mode == 0) {
                    int sel = col / CNL;
                    int head = (col % CNL) / DHEAD;
                    int dd = col % DHEAD;
                    int wi = row / NTOK, p = row % NTOK;
                    if (sel == 0) val *= QSCALE;
                    size_t addr = ((size_t)sel * NWIN * NHEAD +
                                   (size_t)wi * NHEAD + head) * (NTOK * DHEAD) +
                                  p * DHEAD + dd;
                    outB[addr] = f2bf(val);
                } else if (mode == 2) {
                    float g = 0.5f * val * (1.0f + erff(val * 0.70710678118654752f));
                    outB[(size_t)row * N + col] = f2bf(g);
                } else {
                    outF[(size_t)row * N + col] = val;
                }
            }
        }
    }
}

// ---------------------------------------------------------------------------
// Attention: one block per (window, head). 256 threads = 8 waves.
__global__ __launch_bounds__(256)
void attn_kernel(const unsigned short* __restrict__ qkv,
                 const float* __restrict__ bias_table,
                 unsigned short* __restrict__ attnout) {
    const int w = blockIdx.x / NHEAD;
    const int h = blockIdx.x % NHEAD;
    const int tid  = threadIdx.x;
    const int lane = tid & 31;
    const int wv   = tid >> 5;          // wave = M tile (16 query rows)
    const int half = lane >> 4;
    const int l15  = lane & 15;

    __shared__ unsigned short q_lds[128 * 32];   // [query][d]
    __shared__ unsigned short k_lds[128 * 32];   // [key][d]   (== B col-major)
    __shared__ unsigned short v_lds[32 * 128];   // [d][key]   (B col-major)
    __shared__ unsigned short p_lds[128 * 128];  // softmax(P) bf16 [query][key]

    const size_t SEG = (size_t)NWIN * NHEAD * NTOK * DHEAD;
    const size_t base = ((size_t)w * NHEAD + h) * (NTOK * DHEAD);

#if HAVE_ASYNC_LDS
    // q/k rows 0..119 are contiguous 7680B copies
    if (tid < 240) {
        copy32(qkv + base + tid * 16, &q_lds[tid * 16]);
        copy32(qkv + SEG + base + tid * 16, &k_lds[tid * 16]);
    }
    if (tid < 128) {  // zero-pad rows 120..127
        ((unsigned*)q_lds)[1920 + tid] = 0u;
        ((unsigned*)k_lds)[1920 + tid] = 0u;
    }
#else
    for (int e = tid; e < 128 * 32; e += 256) {
        int row = e >> 5, kk = e & 31;
        q_lds[e] = (row < NTOK) ? qkv[base + row * DHEAD + kk] : (unsigned short)0;
        k_lds[e] = (row < NTOK) ? qkv[SEG + base + row * DHEAD + kk] : (unsigned short)0;
    }
#endif
    // v: coalesced 16B row loads + transposed LDS stores
    if (tid < 240) {
        int key = tid >> 1, dg = (tid & 1) * 16;
        Vec8U u0, u1;
        u0.q = *(const uint4*)(qkv + 2 * SEG + base + key * DHEAD + dg);
        u1.q = *(const uint4*)(qkv + 2 * SEG + base + key * DHEAD + dg + 8);
#pragma unroll
        for (int i = 0; i < 8; ++i) v_lds[(dg + i) * 128 + key] = u0.s[i];
#pragma unroll
        for (int i = 0; i < 8; ++i) v_lds[(dg + 8 + i) * 128 + key] = u1.s[i];
    }
    if (tid < 128) {  // zero-pad keys 120..127 across all 32 d-rows
        int dd = tid >> 2, q4 = tid & 3;
        ((unsigned*)v_lds)[dd * 64 + 60 + q4] = 0u;
    }
    ASYNC_WAIT();
    __syncthreads();

    // ---- S = q * k^T  (K = 32, one WMMA per 16x16 tile) ----
    FragU qa;
    {
        int row = wv * 16 + l15;
#pragma unroll
        for (int v = 0; v < 8; ++v) {
            int kk = half * 16 + 2 * v;
            qa.u[v] = *(const unsigned*)&q_lds[row * 32 + kk];
        }
    }
    f32x8 sacc[8];
#pragma unroll
    for (int nt = 0; nt < 8; ++nt) {
        FragU kb;
        int col = nt * 16 + l15;
#pragma unroll
        for (int v = 0; v < 8; ++v)
            kb.u[v] = *(const unsigned*)&k_lds[col * 32 + half * 16 + 2 * v];
        f32x8 z = {};
        sacc[nt] = __builtin_amdgcn_wmma_f32_16x16x32_bf16(
            false, qa.v, false, kb.v, (short)0, z, false, false);
    }

    // ---- bias add + in-register softmax (each wave owns its 16 rows) ----
#pragma unroll
    for (int v = 0; v < 8; ++v) {
        int row = wv * 16 + v + half * 8;     // query index
        bool vrow = row < NTOK;
        float mx = -INFINITY;
#pragma unroll
        for (int nt = 0; nt < 8; ++nt) {
            int col = nt * 16 + l15;          // key index
            float s = sacc[nt][v];
            if (vrow && col < NTOK) {
                int idx = pos_idx(row, col);
                s += bias_table[((size_t)idx * NWIN + w) * NHEAD + h];
            } else {
                s = -INFINITY;
            }
            sacc[nt][v] = s;
            mx = fmaxf(mx, s);
        }
#pragma unroll
        for (int off = 1; off < 16; off <<= 1)
            mx = fmaxf(mx, __shfl_xor(mx, off, 32));
        float sum = 0.0f;
#pragma unroll
        for (int nt = 0; nt < 8; ++nt) {
            float e = vrow ? __expf(sacc[nt][v] - mx) : 0.0f;
            sacc[nt][v] = e;
            sum += e;
        }
#pragma unroll
        for (int off = 1; off < 16; off <<= 1)
            sum += __shfl_xor(sum, off, 32);
        float inv = vrow ? (1.0f / sum) : 0.0f;
#pragma unroll
        for (int nt = 0; nt < 8; ++nt) {
            int col = nt * 16 + l15;
            p_lds[row * 128 + col] = f2bf(sacc[nt][v] * inv);
        }
    }
    __syncthreads();

    // ---- O = P * v  (M=16 rows per wave, N=32, K=128 in 4 steps) ----
    f32x8 oacc[2] = {};
#pragma unroll
    for (int ks = 0; ks < 4; ++ks) {
        int k0 = ks * 32;
        FragU pa;
        int row = wv * 16 + l15;
#pragma unroll
        for (int v = 0; v < 8; ++v) {
            int kk = half * 16 + 2 * v;
            pa.u[v] = *(const unsigned*)&p_lds[row * 128 + k0 + kk];
        }
#pragma unroll
        for (int nt = 0; nt < 2; ++nt) {
            FragU vb;
            int col = nt * 16 + l15;          // d index
#pragma unroll
            for (int v = 0; v < 8; ++v)
                vb.u[v] = *(const unsigned*)&v_lds[col * 128 + k0 + half * 16 + 2 * v];
            oacc[nt] = __builtin_amdgcn_wmma_f32_16x16x32_bf16(
                false, pa.v, false, vb.v, (short)0, oacc[nt], false, false);
        }
    }

    // ---- scatter with the reference's no-transpose reshape ----
#pragma unroll
    for (int nt = 0; nt < 2; ++nt) {
#pragma unroll
        for (int v = 0; v < 8; ++v) {
            int n = wv * 16 + v + half * 8;
            int dd = nt * 16 + l15;
            if (n < NTOK) {
                int np = h * 10 + n / 12;
                int cp = (n % 12) * DHEAD + dd;
                attnout[((size_t)w * NTOK + np) * CNL + cp] = f2bf(oacc[nt][v]);
            }
        }
    }
}

// ---------------------------------------------------------------------------
// Fused (optional window-reverse) + LayerNorm + residual. 384 threads/token.
__global__ __launch_bounds__(384)
void ln_res_kernel(const float* __restrict__ src,
                   const float* __restrict__ shortcut,
                   const float* __restrict__ g,
                   const float* __restrict__ b,
                   float* __restrict__ outF,
                   unsigned short* __restrict__ outB,
                   int permuted) {
    const int t = blockIdx.x;
    const int c = threadIdx.x;
    __shared__ float red[384];

    int r = permuted ? tok_to_win(t) : t;
    float val = src[(size_t)r * CNL + c];

    red[c] = val;
    __syncthreads();
    if (c < 128) red[c] += red[c + 128] + red[c + 256];
    __syncthreads();
    for (int s = 64; s > 0; s >>= 1) {
        if (c < s) red[c] += red[c + s];
        __syncthreads();
    }
    float mean = red[0] * (1.0f / CNL);
    __syncthreads();

    float dv = val - mean;
    red[c] = dv * dv;
    __syncthreads();
    if (c < 128) red[c] += red[c + 128] + red[c + 256];
    __syncthreads();
    for (int s = 64; s > 0; s >>= 1) {
        if (c < s) red[c] += red[c + s];
        __syncthreads();
    }
    float var = red[0] * (1.0f / CNL);

    float y = dv * rsqrtf(var + 1e-5f) * g[c] + b[c] + shortcut[(size_t)t * CNL + c];
    outF[(size_t)t * CNL + c] = y;
    if (outB) outB[(size_t)t * CNL + c] = f2bf(y);
}

// ---------------------------------------------------------------------------
extern "C" void kernel_launch(void* const* d_in, const int* in_sizes, int n_in,
                              void* d_out, int out_size, void* d_ws, size_t ws_size,
                              hipStream_t stream) {
    const float* x          = (const float*)d_in[0];
    const float* wqkv       = (const float*)d_in[1];
    const float* bqkv       = (const float*)d_in[2];
    const float* wproj      = (const float*)d_in[3];
    const float* bproj      = (const float*)d_in[4];
    const float* bias_table = (const float*)d_in[5];
    const float* g1         = (const float*)d_in[6];
    const float* b1         = (const float*)d_in[7];
    const float* g2         = (const float*)d_in[8];
    const float* b2         = (const float*)d_in[9];
    const float* wm1        = (const float*)d_in[10];
    const float* bm1        = (const float*)d_in[11];
    const float* wm2        = (const float*)d_in[12];
    const float* bm2        = (const float*)d_in[13];
    float* out              = (float*)d_out;

    char* ws = (char*)d_ws;
    const size_t XW_B = (size_t)TOK * CNL * 2;      // 53,084,160 B
    const size_t o_wqkv = 0;
    const size_t o_wproj = 884736;
    const size_t o_wm1 = 1179648;
    const size_t o_wm2 = 2359296;
    const size_t o_xw  = 4194304;
    const size_t o_qkv = o_xw + XW_B;
    const size_t o_attn = o_qkv + 3 * XW_B;
    const size_t o_proj = o_attn + XW_B;
    const size_t o_x1  = o_proj + (size_t)TOK * CNL * 4;
    const size_t o_x1b = o_xw;    // reuse (xw dead after QKV gemm)
    const size_t o_h   = o_qkv;   // reuse (qkv+attn dead after proj)
    const size_t o_m   = o_proj;  // reuse (proj dead after first LN)

    unsigned short* wqkvb  = (unsigned short*)(ws + o_wqkv);
    unsigned short* wprojb = (unsigned short*)(ws + o_wproj);
    unsigned short* wm1b   = (unsigned short*)(ws + o_wm1);
    unsigned short* wm2b   = (unsigned short*)(ws + o_wm2);
    unsigned short* xw     = (unsigned short*)(ws + o_xw);
    unsigned short* qkvb   = (unsigned short*)(ws + o_qkv);
    unsigned short* attnb  = (unsigned short*)(ws + o_attn);
    float*          projf  = (float*)(ws + o_proj);
    float*          x1f    = (float*)(ws + o_x1);
    unsigned short* x1b    = (unsigned short*)(ws + o_x1b);
    unsigned short* hb     = (unsigned short*)(ws + o_h);
    float*          mf     = (float*)(ws + o_m);

    // 0) weight conversion + window-ordered bf16 activations
    cvt_bf16_kernel<<<512, 256, 0, stream>>>(wqkv,  wqkvb,  (long)CNL * 3 * CNL);
    cvt_bf16_kernel<<<256, 256, 0, stream>>>(wproj, wprojb, (long)CNL * CNL);
    cvt_bf16_kernel<<<512, 256, 0, stream>>>(wm1,   wm1b,   (long)CNL * 4 * CNL);
    cvt_bf16_kernel<<<512, 256, 0, stream>>>(wm2,   wm2b,   (long)4 * CNL * CNL);
    make_xw_kernel<<<2048, 256, 0, stream>>>(x, xw);

    // 1) QKV: (69120 x 384) @ (384 x 1152), scatter q/k/v (q pre-scaled)
    gemm_bf16_kernel<<<dim3(TOK / 128, 1152 / 64), 256, 0, stream>>>(
        xw, wqkvb, bqkv, nullptr, qkvb, TOK, 1152, CNL, 0);

    // 2) windowed attention, one block per (window, head)
    attn_kernel<<<NWIN * NHEAD, 256, 0, stream>>>(qkvb, bias_table, attnb);

    // 3) proj: (69120 x 384) @ (384 x 384) -> fp32 (window order)
    gemm_bf16_kernel<<<dim3(TOK / 128, CNL / 64), 256, 0, stream>>>(
        attnb, wprojb, bproj, projf, nullptr, TOK, CNL, CNL, 1);

    // 4) x1 = x + LN(window_reverse(proj))
    ln_res_kernel<<<TOK, 384, 0, stream>>>(projf, x, g1, b1, x1f, x1b, 1);

    // 5) MLP1: (69120 x 384) @ (384 x 1536) + GeLU -> bf16
    gemm_bf16_kernel<<<dim3(TOK / 128, 1536 / 64), 256, 0, stream>>>(
        x1b, wm1b, bm1, nullptr, hb, TOK, 1536, CNL, 2);

    // 6) MLP2: (69120 x 1536) @ (1536 x 384) -> fp32
    gemm_bf16_kernel<<<dim3(TOK / 128, CNL / 64), 256, 0, stream>>>(
        hb, wm2b, bm2, mf, nullptr, TOK, CNL, 1536, 3);

    // 7) out = x1 + LN(m)
    ln_res_kernel<<<TOK, 384, 0, stream>>>(mf, x1f, g2, b2, out, nullptr, 0);
}
